// ArSSR_27496380629677
// MI455X (gfx1250) — matrix operand order repaired and checked by reading, not compiled, not run on hardware
//
#include <hip/hip_runtime.h>
#include <hip/hip_bf16.h>

typedef __attribute__((ext_vector_type(16))) _Float16 v16h;
typedef __attribute__((ext_vector_type(8)))  _Float16 v8h;
typedef __attribute__((ext_vector_type(8)))  float    v8f;

#define NVOX   32768            // 32^3
#define NCH    96
#define NPTS   64000            // 40^3
#define NPAIR  (8 * NPTS)       // 8 corner offsets
#define NV     (NCH * NVOX)

__device__ __forceinline__ int clampi(int x, int lo, int hi) {
    return x < lo ? lo : (x > hi ? hi : x);
}

// ---------------- conv3d (SAME, 3x3x3, 1 in-ch, 96 out-ch) for hr and atlas ----
__global__ void k_conv(const float* __restrict__ hr, const float* __restrict__ atlas,
                       const float* __restrict__ Wenc, const float* __restrict__ benc,
                       const float* __restrict__ Wea,  const float* __restrict__ bea,
                       float* __restrict__ axi, float* __restrict__ at) {
    int t = blockIdx.x * blockDim.x + threadIdx.x;
    if (t >= 2 * NV) return;
    int which = t / NV;
    int r = t - which * NV;
    int o = r >> 15, s = r & (NVOX - 1);
    int d = s >> 10, h = (s >> 5) & 31, w = s & 31;
    const float* in = which ? atlas : hr;
    const float* Wt = (which ? Wea : Wenc) + o * 27;
    float acc = which ? bea[o] : benc[o];
    #pragma unroll
    for (int kd = 0; kd < 3; ++kd) {
        int dd = d + kd - 1; if ((unsigned)dd >= 32u) continue;
        #pragma unroll
        for (int kh = 0; kh < 3; ++kh) {
            int hh = h + kh - 1; if ((unsigned)hh >= 32u) continue;
            #pragma unroll
            for (int kw = 0; kw < 3; ++kw) {
                int ww = w + kw - 1; if ((unsigned)ww >= 32u) continue;
                acc += in[dd * 1024 + hh * 32 + ww] * Wt[kd * 9 + kh * 3 + kw];
            }
        }
    }
    (which ? at : axi)[r] = acc;
}

// ---------------- q/k/v projections:  q[s,o] = sum_c at[c,s] * Wq[c,o] + bq[o] --
__global__ void k_qkv(const float* __restrict__ at, const float* __restrict__ axi,
                      const float* __restrict__ Wq, const float* __restrict__ bq,
                      const float* __restrict__ Wk, const float* __restrict__ bk,
                      const float* __restrict__ Wv, const float* __restrict__ bv,
                      float* __restrict__ q, float* __restrict__ k, float* __restrict__ v) {
    int t = blockIdx.x * blockDim.x + threadIdx.x;
    if (t >= 3 * NV) return;
    int which = t / NV;
    int r = t - which * NV;
    int s = r / NCH, o = r - s * NCH;
    const float* src = (which == 0) ? at : axi;
    const float* W   = (which == 0) ? Wq : (which == 1 ? Wk : Wv);
    float acc        = ((which == 0) ? bq : (which == 1 ? bk : bv))[o];
    for (int c = 0; c < NCH; ++c)
        acc += src[c * NVOX + s] * W[c * NCH + o];
    ((which == 0) ? q : (which == 1 ? k : v))[r] = acc;
}

// ---------------- attention logits L[i,j] = sum_s q[s,i]*k[s,j] -----------------
__global__ void k_logits(const float* __restrict__ q, const float* __restrict__ k,
                         float* __restrict__ L) {
    __shared__ float red[256];
    int i = blockIdx.x / NCH, j = blockIdx.x % NCH;
    float acc = 0.f;
    for (int s = threadIdx.x; s < NVOX; s += 256)
        acc += q[s * NCH + i] * k[s * NCH + j];
    red[threadIdx.x] = acc;
    __syncthreads();
    for (int off = 128; off > 0; off >>= 1) {
        if ((int)threadIdx.x < off) red[threadIdx.x] += red[threadIdx.x + off];
        __syncthreads();
    }
    if (threadIdx.x == 0) L[i * NCH + j] = red[0];
}

// ---------------- row softmax over 96 cols --------------------------------------
__global__ void k_softmax(const float* __restrict__ L, float* __restrict__ S) {
    __shared__ float tmp[128];
    int i = blockIdx.x, j = threadIdx.x;  // 128 threads
    float x = (j < NCH) ? L[i * NCH + j] : -3.4e38f;
    tmp[j] = x; __syncthreads();
    for (int off = 64; off > 0; off >>= 1) {
        if (j < off) tmp[j] = fmaxf(tmp[j], tmp[j + off]);
        __syncthreads();
    }
    float m = tmp[0]; __syncthreads();
    float e = (j < NCH) ? __expf(x - m) : 0.f;
    tmp[j] = e; __syncthreads();
    for (int off = 64; off > 0; off >>= 1) {
        if (j < off) tmp[j] += tmp[j + off];
        __syncthreads();
    }
    if (j < NCH) S[i * NCH + j] = e / tmp[0];
}

// ---------------- feat[i,s] = sum_j v[s,j] * S[i,j] -----------------------------
__global__ void k_feat(const float* __restrict__ v, const float* __restrict__ S,
                       float* __restrict__ feat) {
    int t = blockIdx.x * blockDim.x + threadIdx.x;
    if (t >= NV) return;
    int i = t >> 15, s = t & (NVOX - 1);
    const float* vr = v + s * NCH;
    const float* sr = S + i * NCH;
    float acc = 0.f;
    for (int j = 0; j < NCH; ++j) acc += vr[j] * sr[j];
    feat[t] = acc;
}

// ---------------- pack f32 weights into WMMA f16 B-fragment layout --------------
// B (KxN): lane n%16 holds column n; halves j: K = slab*32 + (j>=8 ? 16:0) + (lane>=16 ? 8:0) + (j&7)
// Packed so each lane's 16 halves (32B) are contiguous: one b128x2 load/fragment.
__global__ void k_pack(const float* __restrict__ W, _Float16* __restrict__ dst,
                       int Kreal, int Nreal, int Kpad, int Npad) {
    int e = blockIdx.x * blockDim.x + threadIdx.x;
    int total = Kpad * Npad;
    if (e >= total) return;
    int j = e & 15;
    int lane = (e >> 4) & 31;
    int rest = e >> 9;
    int slabs = Kpad >> 5;
    int ksi = rest % slabs;
    int ngi = rest / slabs;
    int n = ngi * 16 + (lane & 15);
    int k = ksi * 32 + (j & 8) * 2 + ((lane >> 4) & 1) * 8 + (j & 7);
    float val = (k < Kreal && n < Nreal) ? W[k * Nreal + n] : 0.f;
    dst[e] = (_Float16)val;
}

// ---------------- one WMMA layer: src(64x256 f16 LDS) @ Wp -> dst(64x256 f16 LDS)
template <int SLABS, bool RELU>
__device__ __forceinline__ void wmma_layer(const _Float16* src, _Float16* dst,
                                           const _Float16* __restrict__ Wp,
                                           const float* __restrict__ bias,
                                           int lane, int wv) {
    const int mg = wv & 3;              // row group (4 groups of 16 rows)
    const int ngBase = (wv >> 2) * 8;   // 16 col groups split across 2 wave-sets
    const int m = mg * 16 + (lane & 15);
    const int kb = ((lane >> 4) & 1) * 8;
    v16h aF[SLABS];
    #pragma unroll
    for (int ks = 0; ks < SLABS; ++ks) {   // documented 16-bit A 16x32 layout
        v8h lo = *(const v8h*)&src[m * 256 + ks * 32 + kb];
        v8h hi = *(const v8h*)&src[m * 256 + ks * 32 + kb + 16];
        aF[ks] = __builtin_shufflevector(lo, hi, 0, 1, 2, 3, 4, 5, 6, 7,
                                                 8, 9, 10, 11, 12, 13, 14, 15);
    }
    #pragma unroll
    for (int tI = 0; tI < 8; ++tI) {
        const int ngi = ngBase + tI;
        const float bn = bias[ngi * 16 + (lane & 15)];
        v8f acc = {bn, bn, bn, bn, bn, bn, bn, bn};
        #pragma unroll
        for (int ks = 0; ks < SLABS; ++ks) {
            v16h bF = *(const v16h*)&Wp[(size_t)((ngi * SLABS + ks) * 32 + lane) * 16];
            acc = __builtin_amdgcn_wmma_f32_16x16x32_f16(
                false, aF[ks], false, bF, (short)0, acc, false, false);
        }
        const int mrow = mg * 16 + ((lane >> 4) & 1) * 8;
        const int n = ngi * 16 + (lane & 15);
        #pragma unroll
        for (int rr = 0; rr < 8; ++rr) {
            float x = acc[rr];
            if (RELU) x = fmaxf(x, 0.f);
            dst[(mrow + rr) * 256 + n] = (_Float16)x;
        }
    }
}

// ---------------- fused: sampling + posemb -> 4-layer MLP (WMMA) ----------------
__global__ __launch_bounds__(256) void k_mlp(
    const float* __restrict__ hr, const float* __restrict__ coord,
    const float* __restrict__ feat,
    const _Float16* __restrict__ W0p, const float* __restrict__ b0,
    const _Float16* __restrict__ W1p, const float* __restrict__ b1,
    const _Float16* __restrict__ W2p, const float* __restrict__ b2,
    const _Float16* __restrict__ W3p, const float* __restrict__ b3,
    float* __restrict__ predw, float* __restrict__ areaw, float* __restrict__ intenw) {
    __shared__ __align__(32) _Float16 bufA[64 * 256];   // 32 KB
    __shared__ __align__(32) _Float16 bufB[64 * 256];   // 32 KB
    const int t = threadIdx.x;
    const int lane = t & 31;
    const int wv = t >> 5;
    const int pBase = blockIdx.x * 64;

    // ---- stage 1: build the (64 x 160) f16 input tile, area & trilinear inten --
    {
        const int r = t >> 2;          // row 0..63
        const int part = t & 3;        // 4 threads per row
        const int p = pBase + r;
        const int kpt = p % NPTS;
        const int corner = p / NPTS;
        const float vx = (corner & 4) ? 1.f : -1.f;
        const float vy = (corner & 2) ? 1.f : -1.f;
        const float vz = (corner & 1) ? 1.f : -1.f;
        const float cd = coord[kpt];
        const float ch = coord[NPTS + kpt];
        const float cw = coord[2 * NPTS + kpt];
        const float r32 = 1.f / 32.f, eps = 1e-6f;
        const float lo = -1.f + 1e-6f, hi = 1.f - 1e-6f;
        const float c0 = fminf(fmaxf(cd + vx * r32 + eps, lo), hi);
        const float c1 = fminf(fmaxf(ch + vy * r32 + eps, lo), hi);
        const float c2 = fminf(fmaxf(cw + vz * r32 + eps, lo), hi);
        // nearest on the 32-grid (feat / hr)
        const float u0 = ((c0 + 1.f) * 32.f - 1.f) * 0.5f;
        const float u1 = ((c1 + 1.f) * 32.f - 1.f) * 0.5f;
        const float u2 = ((c2 + 1.f) * 32.f - 1.f) * 0.5f;
        const int i0 = clampi((int)rintf(u0), 0, 31);
        const int i1 = clampi((int)rintf(u1), 0, 31);
        const int i2 = clampi((int)rintf(u2), 0, 31);
        const int sid = (i0 << 10) + (i1 << 5) + i2;
        if (part < 3) {                          // cols [32p, 32p+32): q_feat
            const int cb = part * 32;
            for (int cc = 0; cc < 32; ++cc)
                bufA[r * 256 + cb + cc] = (_Float16)feat[(cb + cc) * NVOX + sid];
        } else {                                 // cols 96..159: rel + posemb
            const int j0 = clampi((int)rintf(((c0 + 1.f) * 40.f - 1.f) * 0.5f), 0, 39);
            const int j1 = clampi((int)rintf(((c1 + 1.f) * 40.f - 1.f) * 0.5f), 0, 39);
            const int j2 = clampi((int)rintf(((c2 + 1.f) * 40.f - 1.f) * 0.5f), 0, 39);
            const int qid = j0 * 1600 + j1 * 40 + j2;
            float rel[3];
            rel[0] = (cd - coord[qid]) * 32.f;
            rel[1] = (ch - coord[NPTS + qid]) * 32.f;
            rel[2] = (cw - coord[2 * NPTS + qid]) * 32.f;
            bufA[r * 256 + 96] = (_Float16)rel[0];
            bufA[r * 256 + 97] = (_Float16)rel[1];
            bufA[r * 256 + 98] = (_Float16)rel[2];
            #pragma unroll
            for (int f = 0; f < 10; ++f) {
                const float fr = (float)(1 << f);
                #pragma unroll
                for (int i = 0; i < 3; ++i) {
                    const float a = rel[i] * fr;
                    bufA[r * 256 + 99 + f * 6 + i]     = (_Float16)__sinf(a);
                    bufA[r * 256 + 99 + f * 6 + 3 + i] = (_Float16)__cosf(a);
                }
            }
            bufA[r * 256 + 159] = (_Float16)0.f;  // K padded 159 -> 160
            areaw[p] = fabsf(rel[0] * rel[1] * rel[2]) + 1e-9f;
            // trilinear sample of hr (zeros padding, align_corners=False)
            const float f0 = floorf(u0), f1 = floorf(u1), f2 = floorf(u2);
            const float t0 = u0 - f0, t1 = u1 - f1, t2 = u2 - f2;
            const int a0 = (int)f0, a1 = (int)f1, a2 = (int)f2;
            float acc = 0.f;
            #pragma unroll
            for (int dd = 0; dd < 2; ++dd)
                #pragma unroll
                for (int hh = 0; hh < 2; ++hh)
                    #pragma unroll
                    for (int ww = 0; ww < 2; ++ww) {
                        const int x0 = a0 + dd, x1 = a1 + hh, x2 = a2 + ww;
                        const bool valid = (unsigned)x0 < 32u && (unsigned)x1 < 32u &&
                                           (unsigned)x2 < 32u;
                        const float w = (dd ? t0 : 1.f - t0) * (hh ? t1 : 1.f - t1) *
                                        (ww ? t2 : 1.f - t2);
                        const int y0 = clampi(x0, 0, 31), y1 = clampi(x1, 0, 31),
                                  y2 = clampi(x2, 0, 31);
                        acc += (valid ? w : 0.f) * hr[y0 * 1024 + y1 * 32 + y2];
                    }
            intenw[p] = (acc + 1e-9f) / 8.f;
        }
    }
    __syncthreads();

    // ---- 4 MLP layers on tensor cores ----
    wmma_layer<5, true>(bufA, bufB, W0p, b0, lane, wv);   // 160 -> 256, ReLU
    __syncthreads();
    wmma_layer<8, true>(bufB, bufA, W1p, b1, lane, wv);   // 256 -> 256, ReLU
    __syncthreads();
    wmma_layer<8, true>(bufA, bufB, W2p, b2, lane, wv);   // 256 -> 256, ReLU
    __syncthreads();

    // layer 3: 256 -> 1 (padded N=16); waves 0..3, one 16-row group each
    if (wv < 4) {
        const int mg = wv;
        const int m = mg * 16 + (lane & 15);
        const int kb = ((lane >> 4) & 1) * 8;
        const float bn = ((lane & 15) == 0) ? b3[0] : 0.f;
        v8f acc = {bn, bn, bn, bn, bn, bn, bn, bn};
        #pragma unroll
        for (int ks = 0; ks < 8; ++ks) {
            v8h plo = *(const v8h*)&bufB[m * 256 + ks * 32 + kb];
            v8h phi = *(const v8h*)&bufB[m * 256 + ks * 32 + kb + 16];
            v16h aF = __builtin_shufflevector(plo, phi, 0, 1, 2, 3, 4, 5, 6, 7,
                                                        8, 9, 10, 11, 12, 13, 14, 15);
            v16h bF = *(const v16h*)&W3p[(size_t)(ks * 32 + lane) * 16];
            acc = __builtin_amdgcn_wmma_f32_16x16x32_f16(
                false, aF, false, bF, (short)0, acc, false, false);
        }
        if ((lane & 15) == 0) {   // N == 0 column holds the prediction
            const int mrow = mg * 16 + ((lane >> 4) & 1) * 8;
            #pragma unroll
            for (int rr = 0; rr < 8; ++rr)
                predw[pBase + mrow + rr] = acc[rr];
        }
    }
}

// ---------------- final: out[k] = sum_c inten*pred*area / sum_c area ------------
__global__ void k_final(const float* __restrict__ pred, const float* __restrict__ area,
                        const float* __restrict__ inten, float* __restrict__ out) {
    int k = blockIdx.x * blockDim.x + threadIdx.x;
    if (k >= NPTS) return;
    float num = 0.f, den = 0.f;
    #pragma unroll
    for (int c = 0; c < 8; ++c) {
        const int p = c * NPTS + k;
        const float a = area[p];
        num += inten[p] * pred[p] * a;
        den += a;
    }
    out[k] = num / den;
}

extern "C" void kernel_launch(void* const* d_in, const int* in_sizes, int n_in,
                              void* d_out, int out_size, void* d_ws, size_t ws_size,
                              hipStream_t stream) {
    (void)in_sizes; (void)n_in; (void)out_size; (void)ws_size;
    const float* hr    = (const float*)d_in[0];
    const float* coord = (const float*)d_in[1];
    const float* atlas = (const float*)d_in[2];
    const float* Wenc  = (const float*)d_in[3];
    const float* benc  = (const float*)d_in[4];
    const float* Wea   = (const float*)d_in[5];
    const float* bea   = (const float*)d_in[6];
    const float* Wq = (const float*)d_in[7],  *bq = (const float*)d_in[8];
    const float* Wk = (const float*)d_in[9],  *bk = (const float*)d_in[10];
    const float* Wv = (const float*)d_in[11], *bv = (const float*)d_in[12];
    const float* W0 = (const float*)d_in[13], *b0 = (const float*)d_in[14];
    const float* W1 = (const float*)d_in[15], *b1 = (const float*)d_in[16];
    const float* W2 = (const float*)d_in[17], *b2 = (const float*)d_in[18];
    const float* W3 = (const float*)d_in[19], *b3 = (const float*)d_in[20];

    // workspace layout (~76 MB)
    float* ws   = (float*)d_ws;
    float* axi  = ws;                 // 96*32768
    float* at   = axi + NV;
    float* q    = at  + NV;
    float* kk   = q   + NV;
    float* vv   = kk  + NV;
    float* Lg   = vv  + NV;           // 96*96
    float* Sg   = Lg  + NCH * NCH;
    float* feat = Sg  + NCH * NCH;    // 96*32768
    _Float16* W0p = (_Float16*)(feat + NV);   // 160*256 (32B-aligned offsets)
    _Float16* W1p = W0p + 160 * 256;
    _Float16* W2p = W1p + 256 * 256;
    _Float16* W3p = W2p + 256 * 256;
    // phase-disjoint aliases (q/k/v are dead after feat is built)
    float* predw  = q;                // 512000
    float* areaw  = kk;
    float* intenw = vv;

    k_conv<<<(2 * NV + 255) / 256, 256, 0, stream>>>(hr, atlas, Wenc, benc, Wea, bea,
                                                     axi, at);
    k_qkv<<<(3 * NV + 255) / 256, 256, 0, stream>>>(at, axi, Wq, bq, Wk, bk, Wv, bv,
                                                    q, kk, vv);
    k_logits<<<NCH * NCH, 256, 0, stream>>>(q, kk, Lg);
    k_softmax<<<NCH, 128, 0, stream>>>(Lg, Sg);
    k_feat<<<(NV + 255) / 256, 256, 0, stream>>>(vv, Sg, feat);

    k_pack<<<(160 * 256 + 255) / 256, 256, 0, stream>>>(W0, W0p, 159, 256, 160, 256);
    k_pack<<<(256 * 256 + 255) / 256, 256, 0, stream>>>(W1, W1p, 256, 256, 256, 256);
    k_pack<<<(256 * 256 + 255) / 256, 256, 0, stream>>>(W2, W2p, 256, 256, 256, 256);
    k_pack<<<(256 * 16  + 255) / 256, 256, 0, stream>>>(W3, W3p, 256, 1, 256, 16);

    k_mlp<<<NPAIR / 64, 256, 0, stream>>>(hr, coord, feat,
                                          W0p, b0, W1p, b1, W2p, b2, W3p, b3,
                                          predw, areaw, intenw);
    k_final<<<(NPTS + 255) / 256, 256, 0, stream>>>(predw, areaw, intenw, (float*)d_out);
}
